// MotionPredictor_CVAE_16733192585781
// MI455X (gfx1250) — matrix operand order repaired and checked by reading, not compiled, not run on hardware
//
#include <hip/hip_runtime.h>
#include <hip/hip_bf16.h>
#include <stdint.h>
#include <stddef.h>

// ---------------- problem constants ----------------
#define RNN   2048
#define NZ    64
#define IN_F  69
#define BATCH 256
#define S_ENC 49          // encoder steps (49 pairs incl. dec[0])
#define T_DEC 25
#define G3    (3*RNN)     // 6144
#define K_FC0 (2*IN_F)    // 138
#define K_FC1 (NZ+IN_F)   // 133

typedef __bf16 bf16_t;
typedef __attribute__((ext_vector_type(16))) __bf16 v16bf;
typedef __attribute__((ext_vector_type(8)))  __bf16 v8bf;
typedef __attribute__((ext_vector_type(8)))  float  v8f;

__device__ __forceinline__ bf16_t f2bf(float f) {
  union { float f; uint32_t u; } x; x.f = f;
  uint32_t r = x.u + 0x7FFFu + ((x.u >> 16) & 1u);   // round-to-nearest-even
  uint16_t h = (uint16_t)(r >> 16);
  return __builtin_bit_cast(bf16_t, h);
}
__device__ __forceinline__ float sigmoidf(float x) { return 1.0f / (1.0f + expf(-x)); }

// ---------------- fp32 -> bf16 weight conversion ----------------
__global__ __launch_bounds__(256) void cvt_bf16_kernel(const float* __restrict__ src,
                                                       bf16_t* __restrict__ dst, size_t n) {
  size_t i = (size_t)blockIdx.x * 256 + threadIdx.x;
  if (i < n) dst[i] = f2bf(src[i]);
}

// ---------------- zero h / hb ----------------
__global__ __launch_bounds__(256) void hzero_kernel(float* __restrict__ h, bf16_t* __restrict__ hb) {
  int i = blockIdx.x * 256 + threadIdx.x;
  if (i < BATCH * RNN) { h[i] = 0.0f; hb[i] = f2bf(0.0f); }
}

// ---------------- fc0: pairs @ fc0_w.T -> xb (bf16) for all 49 steps ----------------
// row = s*256 + b ; pair[k] = enc[b,s,k] (k<69), enc[b,s+1,k-69] or dec[b,0,k-69] (k>=69)
__global__ __launch_bounds__(256) void fc0_kernel(const float* __restrict__ enc,
                                                  const float* __restrict__ dec,
                                                  const float* __restrict__ w,   // [2048,138]
                                                  const float* __restrict__ bia, // [2048]
                                                  bf16_t* __restrict__ xb) {     // [49*256, 2048]
  size_t idx = (size_t)blockIdx.x * 256 + threadIdx.x;
  if (idx >= (size_t)S_ENC * BATCH * RNN) return;
  int n   = (int)(idx % RNN);
  int row = (int)(idx / RNN);
  int s = row / BATCH, b = row % BATCH;
  const float* w0 = w + (size_t)n * K_FC0;
  const float* e0 = enc + ((size_t)b * S_ENC + s) * IN_F;
  const float* e1 = (s < S_ENC - 1) ? enc + ((size_t)b * S_ENC + s + 1) * IN_F
                                    : dec + (size_t)b * T_DEC * IN_F;   // dec[b,0,:]
  float acc = bia[n];
  #pragma unroll 3
  for (int k = 0; k < IN_F; ++k) acc = fmaf(e0[k], w0[k], acc);
  #pragma unroll 3
  for (int k = 0; k < IN_F; ++k) acc = fmaf(e1[k], w0[IN_F + k], acc);
  xb[idx] = f2bf(acc);
}

// ---------------- WMMA GEMM: C[M,N] = A[M,K](bf16) @ W[N,K](bf16)^T + bias[N] ----------------
// block = 256 threads = 8 waves stacked along M; wave tile 16(M) x 64(N);
// block tile 128(M) x 64(N). All 8 waves stream identical B fragments -> WGP$ broadcast;
// weight L2 overread is M/128 = 2x. 4 accumulators keeps VGPR pressure low enough for the
// scheduler to multi-buffer fragments (loads overlap WMMA issue instead of waitcnt-0 stalls).
__global__ __launch_bounds__(256)
void wmma_gemm_bf16(const bf16_t* __restrict__ A, const bf16_t* __restrict__ W,
                    const float* __restrict__ bias, float* __restrict__ C,
                    int M, int N, int K) {
  const int wave  = threadIdx.x >> 5;   // 0..7 -> M strip
  const int lane  = threadIdx.x & 31;
  const int half  = lane >> 4;          // 0: lanes 0-15, 1: lanes 16-31
  const int lrow  = lane & 15;
  const int m0 = blockIdx.x * 128 + wave * 16;
  const int n0 = blockIdx.y * 64;

  v8f acc0 = {}, acc1 = {}, acc2 = {}, acc3 = {};

  // A fragment (16-bit A 16x32, ISA 7.12.2): lane half selects K {0..7} vs {8..15},
  // second VGPR quad holds K+16.
  const bf16_t* arow = A + (size_t)(m0 + lrow) * K + half * 8;
  // B fragment (32x16 K-major): lane = column, K {0..7} vs {8..15} by lane half, second quad K+16.
  const bf16_t* wbase = W + (size_t)(n0 + lrow) * K + half * 8;

  for (int k0 = 0; k0 < K; k0 += 32) {
    union { v16bf v; struct { v8bf lo, hi; } p; } ua;
    ua.p.lo = *(const v8bf*)(arow + k0);
    ua.p.hi = *(const v8bf*)(arow + k0 + 16);

    __builtin_prefetch((const void*)(wbase + k0 + 128), 0, 1);  // global_prefetch_b8

    union { v16bf v; struct { v8bf lo, hi; } p; } ub;
    const bf16_t* wr;
    wr = wbase + (size_t)0 * 16 * K + k0;
    ub.p.lo = *(const v8bf*)(wr); ub.p.hi = *(const v8bf*)(wr + 16);
    acc0 = __builtin_amdgcn_wmma_f32_16x16x32_bf16(false, ua.v, false, ub.v, (short)0, acc0, false, false);
    wr = wbase + (size_t)1 * 16 * K + k0;
    ub.p.lo = *(const v8bf*)(wr); ub.p.hi = *(const v8bf*)(wr + 16);
    acc1 = __builtin_amdgcn_wmma_f32_16x16x32_bf16(false, ua.v, false, ub.v, (short)0, acc1, false, false);
    wr = wbase + (size_t)2 * 16 * K + k0;
    ub.p.lo = *(const v8bf*)(wr); ub.p.hi = *(const v8bf*)(wr + 16);
    acc2 = __builtin_amdgcn_wmma_f32_16x16x32_bf16(false, ua.v, false, ub.v, (short)0, acc2, false, false);
    wr = wbase + (size_t)3 * 16 * K + k0;
    ub.p.lo = *(const v8bf*)(wr); ub.p.hi = *(const v8bf*)(wr + 16);
    acc3 = __builtin_amdgcn_wmma_f32_16x16x32_bf16(false, ua.v, false, ub.v, (short)0, acc3, false, false);
  }

  // C/D layout: VGPR j -> row m0 + half*8 + j, col n0 + nt*16 + lrow
  const int rbase = m0 + half * 8;
  #pragma unroll
  for (int nt = 0; nt < 4; ++nt) {
    const v8f* a = (nt == 0) ? &acc0 : (nt == 1) ? &acc1 : (nt == 2) ? &acc2 : &acc3;
    int col = n0 + nt * 16 + lrow;
    float bv = bias ? bias[col] : 0.0f;
    #pragma unroll
    for (int j = 0; j < 8; ++j)
      C[(size_t)(rbase + j) * N + col] = (*a)[j] + bv;
  }
}

// ---------------- fused GRU gate: h' = (1-z)*n + z*h ; also bf16 copy ----------------
__global__ __launch_bounds__(256) void gru_gate_kernel(const float* __restrict__ gi,
                                                       const float* __restrict__ gh,
                                                       float* __restrict__ h,
                                                       bf16_t* __restrict__ hb) {
  int idx = blockIdx.x * 256 + threadIdx.x;          // BATCH*RNN
  if (idx >= BATCH * RNN) return;
  int b = idx / RNN, j = idx % RNN;
  size_t o = (size_t)b * G3 + j;
  float r = sigmoidf(gi[o]            + gh[o]);
  float z = sigmoidf(gi[o + RNN]      + gh[o + RNN]);
  float n = tanhf   (gi[o + 2 * RNN]  + r * gh[o + 2 * RNN]);
  float hv = (1.0f - z) * n + z * h[idx];
  h[idx] = hv;
  hb[idx] = f2bf(hv);
}

// ---------------- latent: mu, logvar -> d_out ; z = mu + eps*exp(0.5*logvar) ----------------
__global__ __launch_bounds__(256) void latent_kernel(const float* __restrict__ h,
                                                     const float* __restrict__ mu_w,
                                                     const float* __restrict__ mu_b,
                                                     const float* __restrict__ sg_w,
                                                     const float* __restrict__ sg_b,
                                                     const float* __restrict__ eps,
                                                     float* __restrict__ out_mu,
                                                     float* __restrict__ out_lv,
                                                     float* __restrict__ z) {
  int idx = blockIdx.x * 256 + threadIdx.x;          // BATCH*NZ
  if (idx >= BATCH * NZ) return;
  int n = idx % NZ, b = idx / NZ;
  const float* hr = h + (size_t)b * RNN;
  const float* mw = mu_w + (size_t)n * RNN;
  const float* sw = sg_w + (size_t)n * RNN;
  float am = mu_b[n], as = sg_b[n];
  for (int k = 0; k < RNN; ++k) { am = fmaf(hr[k], mw[k], am); as = fmaf(hr[k], sw[k], as); }
  out_mu[idx] = am;
  out_lv[idx] = as;
  z[idx] = am + eps[idx] * expf(0.5f * as);
}

// ---------------- fc1: h = cat(z, dec0) @ fc1_w.T + b ----------------
__global__ __launch_bounds__(256) void fc1_kernel(const float* __restrict__ z,
                                                  const float* __restrict__ dec,
                                                  const float* __restrict__ w,   // [2048,133]
                                                  const float* __restrict__ bia,
                                                  float* __restrict__ h,
                                                  bf16_t* __restrict__ hb) {
  int idx = blockIdx.x * 256 + threadIdx.x;          // BATCH*RNN
  if (idx >= BATCH * RNN) return;
  int n = idx % RNN, b = idx / RNN;
  const float* w0 = w + (size_t)n * K_FC1;
  const float* zr = z + (size_t)b * NZ;
  const float* d0 = dec + (size_t)b * T_DEC * IN_F;  // dec[b,0,:]
  float acc = bia[n];
  #pragma unroll 4
  for (int k = 0; k < NZ; ++k)   acc = fmaf(zr[k], w0[k], acc);
  #pragma unroll 3
  for (int k = 0; k < IN_F; ++k) acc = fmaf(d0[k], w0[NZ + k], acc);
  h[idx] = acc;
  hb[idx] = f2bf(acc);
}

// ---------------- init decoder input: inp = dec[:,0,:] ----------------
__global__ __launch_bounds__(256) void init_inp_kernel(const float* __restrict__ dec,
                                                       float* __restrict__ inp) {
  int idx = blockIdx.x * 256 + threadIdx.x;          // BATCH*IN_F
  if (idx >= BATCH * IN_F) return;
  int n = idx % IN_F, b = idx / IN_F;
  inp[idx] = dec[(size_t)b * T_DEC * IN_F + n];
}

// ---------------- decoder gi = inp @ c2_wih.T + b_ih (K=69, plain fp32) ----------------
__global__ __launch_bounds__(256) void dec_gi_kernel(const float* __restrict__ inp,
                                                     const float* __restrict__ w,   // [6144,69]
                                                     const float* __restrict__ bia,
                                                     float* __restrict__ gi) {
  int idx = blockIdx.x * 256 + threadIdx.x;          // BATCH*G3
  if (idx >= BATCH * G3) return;
  int n = idx % G3, b = idx / G3;
  const float* w0 = w + (size_t)n * IN_F;
  const float* x0 = inp + (size_t)b * IN_F;
  float acc = bia[n];
  #pragma unroll 3
  for (int k = 0; k < IN_F; ++k) acc = fmaf(x0[k], w0[k], acc);
  gi[idx] = acc;
}

// ---------------- decoder output: out = inp + h @ fc2_w.T + b ; inp <- out ----------------
__global__ __launch_bounds__(256) void dec_out_kernel(const float* __restrict__ h,
                                                      const float* __restrict__ w,   // [69,2048]
                                                      const float* __restrict__ bia,
                                                      float* __restrict__ inp,       // in/out
                                                      float* __restrict__ out,       // d_out base
                                                      int t) {
  int idx = blockIdx.x * 256 + threadIdx.x;          // BATCH*IN_F
  if (idx >= BATCH * IN_F) return;
  int n = idx % IN_F, b = idx / IN_F;
  const float* hr = h + (size_t)b * RNN;
  const float* w0 = w + (size_t)n * RNN;
  float acc = bia[n];
  for (int k = 0; k < RNN; ++k) acc = fmaf(hr[k], w0[k], acc);
  float v = inp[idx] + acc;
  out[((size_t)b * T_DEC + t) * IN_F + n] = v;
  inp[idx] = v;                                      // feedback for next step
}

// ---------------- host-side orchestration ----------------
extern "C" void kernel_launch(void* const* d_in, const int* in_sizes, int n_in,
                              void* d_out, int out_size, void* d_ws, size_t ws_size,
                              hipStream_t stream) {
  const float* enc    = (const float*)d_in[0];
  const float* dec    = (const float*)d_in[1];
  const float* eps    = (const float*)d_in[2];
  const float* fc0_w  = (const float*)d_in[3];
  const float* fc0_b  = (const float*)d_in[4];
  const float* fc1_w  = (const float*)d_in[5];
  const float* fc1_b  = (const float*)d_in[6];
  const float* fc2_w  = (const float*)d_in[7];
  const float* fc2_b  = (const float*)d_in[8];
  const float* mu_w   = (const float*)d_in[9];
  const float* mu_b   = (const float*)d_in[10];
  const float* sg_w   = (const float*)d_in[11];
  const float* sg_b   = (const float*)d_in[12];
  const float* c1_wih = (const float*)d_in[13];
  const float* c1_whh = (const float*)d_in[14];
  const float* c1_bih = (const float*)d_in[15];
  const float* c1_bhh = (const float*)d_in[16];
  const float* c2_wih = (const float*)d_in[17];
  const float* c2_whh = (const float*)d_in[18];
  const float* c2_bih = (const float*)d_in[19];
  const float* c2_bhh = (const float*)d_in[20];

  float* out = (float*)d_out;
  float* out_mu = out + (size_t)BATCH * T_DEC * IN_F;
  float* out_lv = out_mu + (size_t)BATCH * NZ;

  // workspace carve-up (256B aligned)
  char* ws = (char*)d_ws;
  size_t off = 0;
  auto carve = [&](size_t bytes) -> char* {
    char* p = ws + off;
    off = (off + bytes + 255) & ~(size_t)255;
    return p;
  };
  bf16_t* wih_bf   = (bf16_t*)carve((size_t)G3 * RNN * sizeof(bf16_t));   // 25 MB
  bf16_t* whh_bf   = (bf16_t*)carve((size_t)G3 * RNN * sizeof(bf16_t));   // 25 MB
  bf16_t* c2whh_bf = (bf16_t*)carve((size_t)G3 * RNN * sizeof(bf16_t));   // 25 MB
  bf16_t* xb       = (bf16_t*)carve((size_t)S_ENC * BATCH * RNN * sizeof(bf16_t)); // 51 MB
  float*  h        = (float*) carve((size_t)BATCH * RNN * sizeof(float));
  bf16_t* hb       = (bf16_t*)carve((size_t)BATCH * RNN * sizeof(bf16_t));
  float*  gi       = (float*) carve((size_t)BATCH * G3 * sizeof(float));
  float*  gh       = (float*) carve((size_t)BATCH * G3 * sizeof(float));
  float*  zbuf     = (float*) carve((size_t)BATCH * NZ * sizeof(float));
  float*  inp      = (float*) carve((size_t)BATCH * IN_F * sizeof(float));
  (void)ws_size; (void)n_in; (void)in_sizes; (void)out_size;

  const size_t WN = (size_t)G3 * RNN;                 // 12.58M elems per big weight
  const int cvt_blocks = (int)((WN + 255) / 256);
  cvt_bf16_kernel<<<cvt_blocks, 256, 0, stream>>>(c1_wih, wih_bf, WN);
  cvt_bf16_kernel<<<cvt_blocks, 256, 0, stream>>>(c1_whh, whh_bf, WN);
  cvt_bf16_kernel<<<cvt_blocks, 256, 0, stream>>>(c2_whh, c2whh_bf, WN);

  // fc0 for all 49 steps -> xb (bf16)
  {
    size_t total = (size_t)S_ENC * BATCH * RNN;
    fc0_kernel<<<(int)((total + 255) / 256), 256, 0, stream>>>(enc, dec, fc0_w, fc0_b, xb);
  }

  hzero_kernel<<<(BATCH * RNN + 255) / 256, 256, 0, stream>>>(h, hb);

  const dim3 ggrid(BATCH / 128, G3 / 64);             // 2 x 96 = 192 blocks
  const int gate_blocks = (BATCH * RNN + 255) / 256;

  // -------- encoder recurrence --------
  for (int s = 0; s < S_ENC; ++s) {
    wmma_gemm_bf16<<<ggrid, 256, 0, stream>>>(xb + (size_t)s * BATCH * RNN, wih_bf,
                                              c1_bih, gi, BATCH, G3, RNN);
    wmma_gemm_bf16<<<ggrid, 256, 0, stream>>>(hb, whh_bf, c1_bhh, gh, BATCH, G3, RNN);
    gru_gate_kernel<<<gate_blocks, 256, 0, stream>>>(gi, gh, h, hb);
  }

  // -------- latent --------
  latent_kernel<<<(BATCH * NZ + 255) / 256, 256, 0, stream>>>(h, mu_w, mu_b, sg_w, sg_b,
                                                              eps, out_mu, out_lv, zbuf);
  fc1_kernel<<<gate_blocks, 256, 0, stream>>>(zbuf, dec, fc1_w, fc1_b, h, hb);
  init_inp_kernel<<<(BATCH * IN_F + 255) / 256, 256, 0, stream>>>(dec, inp);

  // -------- decoder recurrence --------
  const int decgi_blocks = (BATCH * G3 + 255) / 256;
  const int out_blocks = (BATCH * IN_F + 255) / 256;
  for (int t = 0; t < T_DEC; ++t) {
    dec_gi_kernel<<<decgi_blocks, 256, 0, stream>>>(inp, c2_wih, c2_bih, gi);
    wmma_gemm_bf16<<<ggrid, 256, 0, stream>>>(hb, c2whh_bf, c2_bhh, gh, BATCH, G3, RNN);
    gru_gate_kernel<<<gate_blocks, 256, 0, stream>>>(gi, gh, h, hb);
    dec_out_kernel<<<out_blocks, 256, 0, stream>>>(h, fc2_w, fc2_b, inp, out, t);
  }
}
